// SingleHeadAttention_16381005267186
// MI455X (gfx1250) — matrix operand order repaired and checked by reading, not compiled, not run on hardware
//
#include <hip/hip_runtime.h>

typedef __attribute__((ext_vector_type(16))) __bf16        v16bf;
typedef __attribute__((ext_vector_type(8)))  float         v8f;
typedef __attribute__((ext_vector_type(4)))  unsigned int  v4u;
typedef __attribute__((ext_vector_type(4)))  int           v4i;

union FragBF { v16bf v; v4u d[2]; };

// ---- feature probes -------------------------------------------------------
#if defined(__has_builtin)
#if __has_builtin(__builtin_amdgcn_global_load_async_to_lds_b128) && \
    __has_builtin(__builtin_amdgcn_s_wait_asynccnt)
#define USE_ASYNC_LDS 1
#endif
#if __has_builtin(__builtin_amdgcn_cvt_pk_bf16_f32)
#define USE_CVT_PK_BF16 1
#endif
#endif

typedef __attribute__((address_space(1))) v4i g_v4i;   // global v4i (builtin param 0)
typedef __attribute__((address_space(3))) v4i l_v4i;   // LDS v4i    (builtin param 1)

__device__ __forceinline__ unsigned short f2bf(float f) {
  unsigned int u = __builtin_bit_cast(unsigned int, f);
  u += 0x7fffu + ((u >> 16) & 1u);          // round-to-nearest-even
  return (unsigned short)(u >> 16);
}

#ifdef USE_CVT_PK_BF16
typedef __attribute__((ext_vector_type(2))) __bf16 v2bf;
__device__ __forceinline__ unsigned int pack_bf16(float a, float b) {
  v2bf p = __builtin_amdgcn_cvt_pk_bf16_f32(a, b);
  return __builtin_bit_cast(unsigned int, p);
}
#else
__device__ __forceinline__ unsigned int pack_bf16(float a, float b) {
  return (unsigned int)f2bf(a) | ((unsigned int)f2bf(b) << 16);
}
#endif

#ifdef USE_ASYNC_LDS
__device__ __forceinline__ void async_cp16(const unsigned short* g, unsigned short* l) {
  __builtin_amdgcn_global_load_async_to_lds_b128((g_v4i*)g, (l_v4i*)l, 0, 0);
}
#endif

// A-fragment (16x32 bf16): lane holds row m=lane%16; v0-3 = K[kofs+8h..+7], v4-7 = K[kofs+16+8h..+7]
__device__ __forceinline__ v16bf ld_frag_a(const unsigned short* base, int row,
                                           int rowlen, int kofs, int lh) {
  FragBF f;
  const unsigned short* p = base + row * rowlen + kofs + 8 * lh;
  f.d[0] = *(const v4u*)(p);
  f.d[1] = *(const v4u*)(p + 16);
  return f.v;
}
// B-fragment (32x16 bf16) from row-major B^T [N][K]: lane holds col n=lane%16; v0-3 = K[kofs+8h..+7], v4-7 = K[kofs+16+8h..+7]
__device__ __forceinline__ v16bf ld_frag_b(const unsigned short* base, int n,
                                           int rowlen, int kofs, int lh) {
  FragBF f;
  const unsigned short* p = base + n * rowlen + kofs + 8 * lh;
  f.d[0] = *(const v4u*)(p);
  f.d[1] = *(const v4u*)(p + 16);
  return f.v;
}

// ---------------------------------------------------------------------------
// Kernel 1: y = (x @ W^T + b) * scale -> bf16. M=16384 (128 rows/WG), N=128, K=2048.
// Software-pipelined: chunk k+1 globals stream into registers during chunk k WMMAs.
// TRANS_OUT=0: row-major out [t][128] (Q, K).  TRANS_OUT=1: transposed out
// [b*128+h][4096] (V^T) with packed 16-byte stores (8 t-consecutive rows/lane).
// ---------------------------------------------------------------------------
template <int TRANS_OUT>
__global__ __launch_bounds__(256, 2)
void qkv_proj_kernel(const float* __restrict__ x, const float* __restrict__ w,
                     const float* __restrict__ bias, unsigned short* __restrict__ out,
                     float scale) {
  __shared__ __align__(16) unsigned short xs[128 * 32];
  __shared__ __align__(16) unsigned short wsm[128 * 32];
  const int tid = threadIdx.x;
  const int wave = tid >> 5, lane = tid & 31;
  const int lm = lane & 15, lh = lane >> 4;
  const size_t m0 = (size_t)blockIdx.x * 128;

  // Per-thread staging slots (4 float4 of x + 4 of W per 32-wide K chunk)
  const int srow[4] = { (tid + 0)   >> 3, (tid + 256) >> 3,
                        (tid + 512) >> 3, (tid + 768) >> 3 };
  const int sc4 = (tid & 7) << 2;

  const v8f vzero = {0.f, 0.f, 0.f, 0.f, 0.f, 0.f, 0.f, 0.f};
  v8f acc[8];
#pragma unroll
  for (int i = 0; i < 8; ++i) acc[i] = vzero;

  float4 xr[4], wr[4];
#pragma unroll
  for (int it = 0; it < 4; ++it) {       // prefetch chunk 0
    xr[it] = *(const float4*)(x + (m0 + srow[it]) * 2048 + sc4);
    wr[it] = *(const float4*)(w + (size_t)srow[it] * 2048 + sc4);
  }

  for (int kc = 0; kc < 2048; kc += 32) {
    __syncthreads();                     // LDS free (previous compute done)
#pragma unroll
    for (int it = 0; it < 4; ++it) {     // convert + stage current chunk
      int row = srow[it];
      *(unsigned int*)(xs  + row * 32 + sc4)     = pack_bf16(xr[it].x, xr[it].y);
      *(unsigned int*)(xs  + row * 32 + sc4 + 2) = pack_bf16(xr[it].z, xr[it].w);
      *(unsigned int*)(wsm + row * 32 + sc4)     = pack_bf16(wr[it].x, wr[it].y);
      *(unsigned int*)(wsm + row * 32 + sc4 + 2) = pack_bf16(wr[it].z, wr[it].w);
    }
    __syncthreads();                     // tile visible
    if (kc + 32 < 2048) {                // issue next chunk's globals now;
#pragma unroll                           // they stream during the WMMAs below
      for (int it = 0; it < 4; ++it) {
        xr[it] = *(const float4*)(x + (m0 + srow[it]) * 2048 + (kc + 32) + sc4);
        wr[it] = *(const float4*)(w + (size_t)srow[it] * 2048 + (kc + 32) + sc4);
      }
    }
    v16bf a = ld_frag_a(xs, wave * 16 + lm, 32, 0, lh);
#pragma unroll
    for (int nt = 0; nt < 8; ++nt) {
      v16bf bb = ld_frag_b(wsm, nt * 16 + lm, 32, 0, lh);
      acc[nt] = __builtin_amdgcn_wmma_f32_16x16x32_bf16(false, a, false, bb,
                                                        (short)0, acc[nt], false, false);
    }
  }
  // Epilogue
  if (TRANS_OUT) {
    const int bidx = (int)(m0 >> 12);                     // batch (tiles never straddle b)
    const int t0 = (int)(m0 & 4095) + wave * 16 + 8 * lh; // r=0..7 are consecutive t
#pragma unroll
    for (int nt = 0; nt < 8; ++nt) {
      int n = nt * 16 + lm;
      float bv = bias[n];
      v4u pk;
      pk.x = pack_bf16(acc[nt][0] + bv, acc[nt][1] + bv);
      pk.y = pack_bf16(acc[nt][2] + bv, acc[nt][3] + bv);
      pk.z = pack_bf16(acc[nt][4] + bv, acc[nt][5] + bv);
      pk.w = pack_bf16(acc[nt][6] + bv, acc[nt][7] + bv);
      *(v4u*)(out + ((size_t)(bidx * 128 + n)) * 4096 + t0) = pk;
    }
  } else {
#pragma unroll
    for (int nt = 0; nt < 8; ++nt) {
      int n = nt * 16 + lm;
      float bv = bias[n];
#pragma unroll
      for (int r = 0; r < 8; ++r) {
        size_t row = m0 + wave * 16 + r + 8 * lh;
        out[row * 128 + n] = f2bf((acc[nt][r] + bv) * scale);
      }
    }
  }
}

// ---------------------------------------------------------------------------
// Kernel 2: causal flash attention. Br=Bc=64, 128 threads (4 waves x 16 q-rows).
// Q pre-scaled by 1/sqrt(128). K/V^T tiles double-buffered via async-to-LDS.
// ---------------------------------------------------------------------------
__global__ __launch_bounds__(128, 2)
void attn_kernel(const unsigned short* __restrict__ Qb,
                 const unsigned short* __restrict__ Kb,
                 const unsigned short* __restrict__ Vt,   // [b*128+h][4096] bf16
                 float* __restrict__ out) {
  __shared__ __align__(16) unsigned short qs[64 * 128];       // Q tile
  __shared__ __align__(16) unsigned short ks[2][64 * 128];    // K tile (dbl buf)
  __shared__ __align__(16) unsigned short vs[2][128 * 64];    // V^T tile (dbl buf)
  __shared__ __align__(16) unsigned short ps[64 * 64];        // P (per-wave slabs)

  const int tid = threadIdx.x;
  const int wave = tid >> 5, lane = tid & 31;
  const int lm = lane & 15, lh = lane >> 4;
  const int b = blockIdx.y, qt = blockIdx.x;
  const size_t rowbase = (size_t)b * 4096;
  const size_t vtbase = (size_t)b * 128 * 4096;

  // --- tile loaders: 1024 x 16B chunks, 8 per thread ---
  auto load_kv = [&](int j, int buf) {
    const unsigned short* ksrc = Kb + (rowbase + j * 64) * 128;   // contiguous 16 KB
#pragma unroll
    for (int i = 0; i < 8; ++i) {
      int idx = (tid + i * 128) * 8;
#ifdef USE_ASYNC_LDS
      async_cp16(ksrc + idx, &ks[buf][idx]);
#else
      *(v4u*)(&ks[buf][idx]) = *(const v4u*)(ksrc + idx);
#endif
    }
#pragma unroll
    for (int i = 0; i < 8; ++i) {
      int idx = tid + i * 128;            // (h, 8-elem chunk c)
      int h = idx >> 3, c = (idx & 7) * 8;
      const unsigned short* vsrc = Vt + vtbase + (size_t)h * 4096 + j * 64 + c;
#ifdef USE_ASYNC_LDS
      async_cp16(vsrc, &vs[buf][h * 64 + c]);
#else
      *(v4u*)(&vs[buf][h * 64 + c]) = *(const v4u*)(vsrc);
#endif
    }
  };

  {  // Q tile (contiguous 16 KB)
    const unsigned short* qsrc = Qb + (rowbase + qt * 64) * 128;
#pragma unroll
    for (int i = 0; i < 8; ++i) {
      int idx = (tid + i * 128) * 8;
#ifdef USE_ASYNC_LDS
      async_cp16(qsrc + idx, &qs[idx]);
#else
      *(v4u*)(&qs[idx]) = *(const v4u*)(qsrc + idx);
#endif
    }
  }
  load_kv(0, 0);   // prime tile 0

  const v8f vzero = {0.f, 0.f, 0.f, 0.f, 0.f, 0.f, 0.f, 0.f};
  v8f o[8];
  float mrow[8], lrow[8];
#pragma unroll
  for (int i = 0; i < 8; ++i) { o[i] = vzero; mrow[i] = -1.0e30f; lrow[i] = 0.f; }

  for (int j = 0; j <= qt; ++j) {
    const int buf = j & 1;
    __syncthreads();                       // all waves done with buf^1 (tile j-1)
    if (j < qt) load_kv(j + 1, buf ^ 1);   // stream next tile during compute
#ifdef USE_ASYNC_LDS
    if (j < qt) __builtin_amdgcn_s_wait_asynccnt(16);  // tile j resident, j+1 in flight
    else        __builtin_amdgcn_s_wait_asynccnt(0);
#endif
    __syncthreads();                       // tile j visible to all waves

    // S = Q K^T : 16x64 per wave, K-dim = 128
    v8f s[4];
#pragma unroll
    for (int i = 0; i < 4; ++i) s[i] = vzero;
#pragma unroll
    for (int c = 0; c < 4; ++c) {
      v16bf a = ld_frag_a(qs, wave * 16 + lm, 128, c * 32, lh);
#pragma unroll
      for (int nt = 0; nt < 4; ++nt) {
        v16bf bb = ld_frag_b(ks[buf], nt * 16 + lm, 128, c * 32, lh);
        s[nt] = __builtin_amdgcn_wmma_f32_16x16x32_bf16(false, a, false, bb,
                                                        (short)0, s[nt], false, false);
      }
    }
    // Causal mask on the diagonal tile (additive -1e9, matches reference)
    if (j == qt) {
#pragma unroll
      for (int nt = 0; nt < 4; ++nt) {
        int kl = nt * 16 + lm;
#pragma unroll
        for (int r = 0; r < 8; ++r) {
          int ql = wave * 16 + r + 8 * lh;
          if (kl > ql) s[nt][r] += -1.0e9f;
        }
      }
    }
    // Online softmax per q-row (16-lane half reductions)
#pragma unroll
    for (int r = 0; r < 8; ++r) {
      float mx = s[0][r];
#pragma unroll
      for (int nt = 1; nt < 4; ++nt) mx = fmaxf(mx, s[nt][r]);
#pragma unroll
      for (int d = 1; d < 16; d <<= 1) mx = fmaxf(mx, __shfl_xor(mx, d));
      float mn = fmaxf(mrow[r], mx);
      float alpha = __expf(mrow[r] - mn);
      float rs = 0.f;
#pragma unroll
      for (int nt = 0; nt < 4; ++nt) {
        float p = __expf(s[nt][r] - mn);
        s[nt][r] = p;
        rs += p;
      }
#pragma unroll
      for (int d = 1; d < 16; d <<= 1) rs += __shfl_xor(rs, d);
      lrow[r] = lrow[r] * alpha + rs;
      mrow[r] = mn;
#pragma unroll
      for (int nt = 0; nt < 8; ++nt) o[nt][r] *= alpha;
      int prow = wave * 16 + r + 8 * lh;   // wave-private 16-row slab
#pragma unroll
      for (int nt = 0; nt < 4; ++nt) ps[prow * 64 + nt * 16 + lm] = f2bf(s[nt][r]);
    }
    asm volatile("s_wait_dscnt 0" ::: "memory");  // same-wave LDS RAW

    // O += P V : A = P (16x64), B^T = V^T tile (rowlen 64)
#pragma unroll
    for (int c = 0; c < 2; ++c) {
      v16bf a = ld_frag_a(ps, wave * 16 + lm, 64, c * 32, lh);
#pragma unroll
      for (int nt = 0; nt < 8; ++nt) {
        v16bf bb = ld_frag_b(vs[buf], nt * 16 + lm, 64, c * 32, lh);
        o[nt] = __builtin_amdgcn_wmma_f32_16x16x32_bf16(false, a, false, bb,
                                                        (short)0, o[nt], false, false);
      }
    }
  }
  // Finalize: O / l, fp32 store
#pragma unroll
  for (int r = 0; r < 8; ++r) {
    float inv = 1.0f / lrow[r];
    size_t row = rowbase + qt * 64 + wave * 16 + r + 8 * lh;
    float* op = out + row * 128;
#pragma unroll
    for (int nt = 0; nt < 8; ++nt) op[nt * 16 + lm] = o[nt][r] * inv;
  }
}

extern "C" void kernel_launch(void* const* d_in, const int* in_sizes, int n_in,
                              void* d_out, int out_size, void* d_ws, size_t ws_size,
                              hipStream_t stream) {
  const float* x  = (const float*)d_in[0];
  const float* qw = (const float*)d_in[1];
  const float* qb = (const float*)d_in[2];
  const float* kw = (const float*)d_in[3];
  const float* kb = (const float*)d_in[4];
  const float* vw = (const float*)d_in[5];
  const float* vb = (const float*)d_in[6];

  // Workspace: Q/K row-major bf16, V transposed bf16 -> 12 MB total (L2-resident)
  unsigned short* Q  = (unsigned short*)d_ws;
  unsigned short* K  = Q + (size_t)16384 * 128;
  unsigned short* Vt = K + (size_t)16384 * 128;

  const float qscale = 0.08838834764831845f;  // 1/sqrt(128), folded into Q
  qkv_proj_kernel<0><<<dim3(128), 256, 0, stream>>>(x, qw, qb, Q, qscale);
  qkv_proj_kernel<0><<<dim3(128), 256, 0, stream>>>(x, kw, kb, K, 1.0f);
  qkv_proj_kernel<1><<<dim3(128), 256, 0, stream>>>(x, vw, vb, Vt, 1.0f);
  attn_kernel<<<dim3(64, 4), 128, 0, stream>>>(Q, K, Vt, (float*)d_out);
}
